// Decode_87058987090520
// MI455X (gfx1250) — matrix operand order repaired and checked by reading, not compiled, not run on hardware
//
#include <hip/hip_runtime.h>

typedef __attribute__((ext_vector_type(16))) __bf16 v16bf;
typedef __attribute__((ext_vector_type(8)))  float  v8f;

#define HW      65536
#define WIDTH   256
#define NCAT    80
#define KTOP    100
#define NPT     128
#define SEG     10240
#define NSEG    512

__device__ __forceinline__ unsigned short f2bf(float f) {
  unsigned u = __float_as_uint(f);
  unsigned r = u + 0x7FFFu + ((u >> 16) & 1u);
  return (unsigned short)(r >> 16);
}

union Frag16 { v16bf v; unsigned u[8]; };

// async global->LDS copy of 16B per lane (one wave op = 512B)
__device__ __forceinline__ void async_copy16(unsigned lds, const void* g) {
  asm volatile("global_load_async_to_lds_b128 %0, %1, off"
               :: "v"(lds), "v"((unsigned long long)g) : "memory");
}
__device__ __forceinline__ void async_copy16_off16(unsigned lds, const void* g) {
  asm volatile("global_load_async_to_lds_b128 %0, %1, off offset:16"
               :: "v"(lds), "v"((unsigned long long)g) : "memory");
}
__device__ __forceinline__ void wait_async0() {
  asm volatile("s_wait_asynccnt 0x0" ::: "memory");
}

// ---------------- elementwise f32 -> bf16 (x4 vectorized) ----------------
__global__ void cvt_bf16_kernel(const float4* __restrict__ in,
                                uint2* __restrict__ out, int n4) {
  int i = blockIdx.x * blockDim.x + threadIdx.x;
  if (i < n4) {
    float4 f = in[i];
    uint2 o;
    o.x = (unsigned)f2bf(f.x) | ((unsigned)f2bf(f.y) << 16);
    o.y = (unsigned)f2bf(f.z) | ((unsigned)f2bf(f.w) << 16);
    out[i] = o;
  }
}

// ---------------- sigmoid + 3x3 NMS ----------------
__global__ void nms_kernel(const float* __restrict__ hm, float* __restrict__ scores) {
  int i = blockIdx.x * blockDim.x + threadIdx.x;
  if (i >= NCAT * HW) return;
  int c = i >> 16;
  int hw = i & (HW - 1);
  int y = hw >> 8, x = hw & 255;
  const float* base = hm + c * HW;
  float v = base[hw];
  float mx = v;
  for (int dy = -1; dy <= 1; ++dy) {
    int yy = y + dy; if (yy < 0 || yy >= WIDTH) continue;
    for (int dx = -1; dx <= 1; ++dx) {
      int xx = x + dx; if (xx < 0 || xx >= WIDTH) continue;
      mx = fmaxf(mx, base[yy * WIDTH + xx]);
    }
  }
  scores[i] = (v == mx) ? (1.f / (1.f + __expf(-v))) : 0.f;
}

// ---------------- top-100: segment maxima ----------------
__global__ void topk_stage1(const float* __restrict__ scores,
                            unsigned long long* __restrict__ part) {
  __shared__ unsigned long long sm[256];
  int b = blockIdx.x, tid = threadIdx.x;
  unsigned start = (unsigned)b * SEG;
  unsigned long long best = 0ull;
  for (int i = tid; i < SEG; i += 256) {
    unsigned g = start + i;
    unsigned long long key = ((unsigned long long)__float_as_uint(scores[g]) << 32) | g;
    if (key > best) best = key;
  }
  sm[tid] = best; __syncthreads();
  for (int s = 128; s > 0; s >>= 1) {
    if (tid < s) { unsigned long long o = sm[tid + s]; if (o > sm[tid]) sm[tid] = o; }
    __syncthreads();
  }
  if (tid == 0) part[b] = sm[0];
}

// ---------------- top-100: 100 argmax rounds (single block) ----------------
__global__ void topk_stage2(float* __restrict__ scores,
                            unsigned long long* __restrict__ part,
                            unsigned long long* __restrict__ topk) {
  __shared__ unsigned long long sm[NSEG];
  __shared__ unsigned long long winner;
  int tid = threadIdx.x;
  for (int r = 0; r < KTOP; ++r) {
    sm[tid] = part[tid];
    __syncthreads();
    for (int s = 256; s > 0; s >>= 1) {
      if (tid < s) { unsigned long long o = sm[tid + s]; if (o > sm[tid]) sm[tid] = o; }
      __syncthreads();
    }
    if (tid == 0) {
      winner = sm[0];
      topk[r] = sm[0];
      scores[(unsigned)sm[0]] = 0.f;   // remove winner
    }
    __syncthreads();
    unsigned idx = (unsigned)winner;
    int seg = idx / SEG;
    unsigned long long best = 0ull;
    for (int i = tid; i < SEG; i += NSEG) {
      unsigned g = (unsigned)seg * SEG + i;
      unsigned long long key = ((unsigned long long)__float_as_uint(scores[g]) << 32) | g;
      if (key > best) best = key;
    }
    sm[tid] = best; __syncthreads();
    for (int s = 256; s > 0; s >>= 1) {
      if (tid < s) { unsigned long long o = sm[tid + s]; if (o > sm[tid]) sm[tid] = o; }
      __syncthreads();
    }
    if (tid == 0) part[seg] = sm[0];
    __syncthreads();
  }
}

// ---------------- gather: detections, init polys, points ----------------
__global__ void gather_kernel(const unsigned long long* __restrict__ topk,
                              const float* __restrict__ wh,
                              float* __restrict__ initPolys,
                              float* __restrict__ points,
                              float* __restrict__ validArr,
                              float* __restrict__ dOut) {
  int k = blockIdx.x, p = threadIdx.x;
  unsigned long long key = topk[k];
  float score = __uint_as_float((unsigned)(key >> 32));
  unsigned flat = (unsigned)key;
  int cls = flat >> 16;
  int hw = flat & 0xFFFF;
  float ys = (float)(hw >> 8), xs = (float)(hw & 255);
  float vm = (score >= 0.05f) ? 1.f : 0.f;
  float whx = wh[(2 * p) * HW + hw];
  float why = wh[(2 * p + 1) * HW + hw];
  float ipx = fminf(fmaxf(xs + whx * 10.f, 0.f), 255.f);
  float ipy = fminf(fmaxf(ys + why * 10.f, 0.f), 255.f);
  initPolys[k * 256 + p * 2 + 0] = ipx;
  initPolys[k * 256 + p * 2 + 1] = ipy;
  points[k * 258 + (p + 1) * 2 + 0] = ipx;
  points[k * 258 + (p + 1) * 2 + 1] = ipy;
  dOut[(k * NPT + p) * 2 + 0] = ipx * 4.f * vm;
  dOut[(k * NPT + p) * 2 + 1] = ipy * 4.f * vm;
  if (p == 0) {
    points[k * 258 + 0] = xs;
    points[k * 258 + 1] = ys;
    validArr[k] = vm;
    dOut[51600 + k] = vm;
    dOut[51200 + k * 4 + 0] = xs * vm;
    dOut[51200 + k * 4 + 1] = ys * vm;
    dOut[51200 + k * 4 + 2] = score * vm;
    dOut[51200 + k * 4 + 3] = (float)cls * vm;
  }
}

// ---------------- bilinear grid sample -> bf16 A-matrix ----------------
__global__ void sample_kernel(const float* __restrict__ featmap,
                              const float* __restrict__ points,
                              unsigned short* __restrict__ featA) {
  int t = blockIdx.x;                 // 0 .. 100*129-1
  int k = t / 129, pidx = t - k * 129;
  int c = threadIdx.x;                // 0..63
  float px = points[k * 258 + pidx * 2 + 0];
  float py = points[k * 258 + pidx * 2 + 1];
  // gx = px/128 - 1  ->  x = ((gx+1)*256 - 1)*0.5 = px - 0.5 (same for y)
  float x = px - 0.5f, y = py - 0.5f;
  float x0 = floorf(x), y0 = floorf(y);
  float wx1 = x - x0, wx0 = 1.f - wx1;
  float wy1 = y - y0, wy0 = 1.f - wy1;
  int ix0 = (int)x0, iy0 = (int)y0;
  int ix1 = ix0 + 1, iy1 = iy0 + 1;
  float acc = 0.f;
  {
    float w = ((ix0 >= 0 && ix0 < WIDTH && iy0 >= 0 && iy0 < WIDTH) ? 1.f : 0.f) * wx0 * wy0;
    int idx = min(max(iy0, 0), 255) * WIDTH + min(max(ix0, 0), 255);
    acc += w * featmap[idx * 64 + c];
  }
  {
    float w = ((ix1 >= 0 && ix1 < WIDTH && iy0 >= 0 && iy0 < WIDTH) ? 1.f : 0.f) * wx1 * wy0;
    int idx = min(max(iy0, 0), 255) * WIDTH + min(max(ix1, 0), 255);
    acc += w * featmap[idx * 64 + c];
  }
  {
    float w = ((ix0 >= 0 && ix0 < WIDTH && iy1 >= 0 && iy1 < WIDTH) ? 1.f : 0.f) * wx0 * wy1;
    int idx = min(max(iy1, 0), 255) * WIDTH + min(max(ix0, 0), 255);
    acc += w * featmap[idx * 64 + c];
  }
  {
    float w = ((ix1 >= 0 && ix1 < WIDTH && iy1 >= 0 && iy1 < WIDTH) ? 1.f : 0.f) * wx1 * wy1;
    int idx = min(max(iy1, 0), 255) * WIDTH + min(max(ix1, 0), 255);
    acc += w * featmap[idx * 64 + c];
  }
  featA[k * 8256 + c * 129 + pidx] = f2bf(acc);
}

// ---------------- generic bf16 WMMA GEMM (64x64 tile, 4 waves) ----------------
// Double-buffered LDS, async global->LDS staging, one barrier per K-chunk,
// pipeline unrolled x2 so buffer indices are compile-time (K/32 always even).
// MODE 1: A = implicit im2col of cnn_feature (3x3 pad 1), epilogue bias+relu -> bf16
// MODE 2: A = bf16 row-major,                 epilogue bias -> f32
// MODE 0: A = bf16 row-major,                 epilogue raw  -> bf16
// MODE 3: A = bf16 row-major,                 epilogue coarse-poly -> d_out
#define TM 64
#define TN 64
#define TK 32
#define RS 40   // LDS row stride (u16) -> 80B rows, 16B aligned

template <int MODE>
__global__ __launch_bounds__(128) void gemm_wmma_kernel(
    const float* __restrict__ imsrc,
    const unsigned short* __restrict__ A,
    const unsigned short* __restrict__ Bt,     // [N][K] bf16
    const float* __restrict__ bias,
    unsigned short* __restrict__ outBf,
    float* __restrict__ outF,
    const float* __restrict__ initPolys,
    const float* __restrict__ validArr,
    float* __restrict__ dOut,
    int Mreal, int N, int K) {
  __shared__ unsigned short As[2][TM * RS];
  __shared__ unsigned short Bs[2][TN * RS];
  int tid = threadIdx.x;
  int lane = tid & 31, wave = tid >> 5;
  int m0 = blockIdx.x * TM;
  int n0 = blockIdx.y * TN;

  // staging assignment: thread -> one contiguous 32B run of one tile row
  int row  = tid >> 1;                 // 0..63
  int colh = (tid & 1) << 4;           // 0 or 16 (u16 units)
  unsigned ldsA[2] = { (unsigned)(unsigned long long)&As[0][row * RS + colh],
                       (unsigned)(unsigned long long)&As[1][row * RS + colh] };
  unsigned ldsB[2] = { (unsigned)(unsigned long long)&Bs[0][row * RS + colh],
                       (unsigned)(unsigned long long)&Bs[1][row * RS + colh] };
  // clamp tail rows: they only feed D rows >= Mreal which are never stored
  int mclamp = min(m0 + row, Mreal - 1);
  const unsigned short* aRow = A + (unsigned long long)mclamp * K + colh;
  const unsigned short* bRow = Bt + (unsigned long long)(n0 + row) * K + colh;
  int pixp = m0 + row;
  int yp = pixp >> 8, xp = pixp & 255;

  auto stageAB = [&](int kb, int buf) {
    if (MODE == 1) {
      unsigned short* dst = &As[buf][row * RS + colh];
      // decompose first k once; derive the 15 following incrementally
      int kbase = kb + colh;
      int ci0 = kbase / 9;
      int r0 = kbase - ci0 * 9;        // 0..8
#pragma unroll
      for (int i = 0; i < 16; ++i) {
        int r = r0 + i;                // 0..23 -> at most 2 wraps of 9
        int wrap = (r >= 9 ? 1 : 0) + (r >= 18 ? 1 : 0);
        int ci = ci0 + wrap;
        int rr = r - 9 * wrap;         // 0..8
        int ky = (rr >= 3 ? 1 : 0) + (rr >= 6 ? 1 : 0);
        int kx = rr - 3 * ky;
        int yy = yp + ky - 1, xx = xp + kx - 1;
        float ok = ((unsigned)yy < (unsigned)WIDTH && (unsigned)xx < (unsigned)WIDTH) ? 1.f : 0.f;
        int cy = min(max(yy, 0), 255), cx = min(max(xx, 0), 255);
        dst[i] = f2bf(imsrc[ci * HW + cy * WIDTH + cx] * ok);
      }
    } else {
      async_copy16(ldsA[buf], aRow + kb);
      async_copy16_off16(ldsA[buf], aRow + kb);
    }
    async_copy16(ldsB[buf], bRow + kb);
    async_copy16_off16(ldsB[buf], bRow + kb);
  };

  v8f acc[4] = {};
  int mloc = wave * 16 + (lane & 15);
  int ao = (lane >> 4) * 4;            // A frag u32 offset 0 or 4
  int kh = (lane >> 4) * 4;            // B frag u32 offset 0 or 4

  auto compute = [&](int buf) {
    // A fragment: lanes 0-15 -> K 0-7 & 16-23 ; lanes 16-31 -> K 8-15 & 24-31
    Frag16 af;
    const unsigned* pa = (const unsigned*)&As[buf][mloc * RS];
#pragma unroll
    for (int i = 0; i < 4; ++i) { af.u[i] = pa[ao + i]; af.u[4 + i] = pa[8 + ao + i]; }
    // load all 4 B fragments first, then 4 back-to-back WMMAs
    Frag16 bv[4];
#pragma unroll
    for (int j = 0; j < 4; ++j) {
      const unsigned* pb = (const unsigned*)&Bs[buf][(j * 16 + (lane & 15)) * RS];
#pragma unroll
      for (int i = 0; i < 4; ++i) { bv[j].u[i] = pb[kh + i]; bv[j].u[4 + i] = pb[8 + kh + i]; }
    }
#pragma unroll
    for (int j = 0; j < 4; ++j)
      acc[j] = __builtin_amdgcn_wmma_f32_16x16x32_bf16(
          false, af.v, false, bv[j].v, (short)0, acc[j], false, false);
  };

  int nk = K >> 5;                     // always even for our shapes
  stageAB(0, 0);
  for (int ik = 0; ik < nk; ik += 2) {
    wait_async0();
    __syncthreads();                   // chunk ik resident in buf0
    if (ik + 1 < nk) stageAB((ik + 1) << 5, 1);
    compute(0);
    wait_async0();
    __syncthreads();                   // chunk ik+1 resident in buf1
    if (ik + 2 < nk) stageAB((ik + 2) << 5, 0);
    compute(1);
  }
  // epilogue (C/D layout: lane<16: N=lane, M=vgpr ; lane>=16: N=lane-16, M=vgpr+8)
#pragma unroll
  for (int j = 0; j < 4; ++j) {
#pragma unroll
    for (int vv = 0; vv < 8; ++vv) {
      int Ml = vv + ((lane >> 4) << 3);
      int Nl = lane & 15;
      int mg = m0 + wave * 16 + Ml;
      int ng = n0 + j * 16 + Nl;
      if (mg >= Mreal) continue;
      float val = acc[j][vv];
      if (MODE == 1) {
        outBf[(long long)mg * N + ng] = f2bf(fmaxf(val + bias[ng], 0.f));
      } else if (MODE == 2) {
        outF[(long long)mg * N + ng] = val + bias[ng];
      } else if (MODE == 0) {
        outBf[(long long)mg * N + ng] = f2bf(val);
      } else {  // MODE 3: coarse = clip(off*4 + init_poly) * 4 * valid
        float ip = initPolys[mg * 256 + ng];
        float cz = (val + bias[ng]) * 4.f + ip;
        cz = fminf(fmaxf(cz, 0.f), 255.f);
        dOut[25600 + mg * 256 + ng] = cz * 4.f * validArr[mg];
      }
    }
  }
}

extern "C" void kernel_launch(void* const* d_in, const int* in_sizes, int n_in,
                              void* d_out, int out_size, void* d_ws, size_t ws_size,
                              hipStream_t stream) {
  (void)in_sizes; (void)n_in; (void)out_size; (void)ws_size;
  const float* hm    = (const float*)d_in[0];
  const float* wh    = (const float*)d_in[1];
  const float* cf    = (const float*)d_in[2];
  const float* c1w   = (const float*)d_in[3];
  const float* c1b   = (const float*)d_in[4];
  const float* c2w   = (const float*)d_in[5];
  const float* c2b   = (const float*)d_in[6];
  const float* wpoly = (const float*)d_in[7];
  const float* wfuse = (const float*)d_in[8];
  const float* bfuse = (const float*)d_in[9];
  float* out = (float*)d_out;
  char* ws = (char*)d_ws;
  size_t off = 0;
  auto take = [&](size_t bytes) {
    char* p = ws + off;
    off = (off + bytes + 255) & ~(size_t)255;
    return p;
  };
  float*              scores  = (float*)take(5242880ull * 4);
  unsigned long long* part    = (unsigned long long*)take(512 * 8);
  unsigned long long* topk    = (unsigned long long*)take(100 * 8);
  unsigned short*     bt1     = (unsigned short*)take(256 * 576 * 2);
  unsigned short*     bt2     = (unsigned short*)take(64 * 256 * 2);
  unsigned short*     wpolyB  = (unsigned short*)take(512ull * 8256 * 2);
  unsigned short*     wfuseB  = (unsigned short*)take(256 * 512 * 2);
  unsigned short*     fbf     = (unsigned short*)take(65536ull * 256 * 2);
  float*              featmap = (float*)take(65536ull * 64 * 4);
  unsigned short*     featA   = (unsigned short*)take(100ull * 8256 * 2);
  unsigned short*     t1      = (unsigned short*)take(100 * 512 * 2);
  float*              ipoly   = (float*)take(100 * 256 * 4);
  float*              pts     = (float*)take(100 * 258 * 4);
  float*              validA  = (float*)take(100 * 4);

  // weight conversions (layouts already [N][K]); sizes all divisible by 4
  cvt_bf16_kernel<<<(147456 / 4 + 255) / 256, 256, 0, stream>>>(
      (const float4*)c1w, (uint2*)bt1, 147456 / 4);
  cvt_bf16_kernel<<<(16384 / 4 + 255) / 256, 256, 0, stream>>>(
      (const float4*)c2w, (uint2*)bt2, 16384 / 4);
  cvt_bf16_kernel<<<(4227072 / 4 + 255) / 256, 256, 0, stream>>>(
      (const float4*)wpoly, (uint2*)wpolyB, 4227072 / 4);
  cvt_bf16_kernel<<<(131072 / 4 + 255) / 256, 256, 0, stream>>>(
      (const float4*)wfuse, (uint2*)wfuseB, 131072 / 4);

  nms_kernel<<<20480, 256, 0, stream>>>(hm, scores);
  topk_stage1<<<512, 256, 0, stream>>>(scores, part);
  topk_stage2<<<1, NSEG, 0, stream>>>(scores, part, topk);
  gather_kernel<<<KTOP, NPT, 0, stream>>>(topk, wh, ipoly, pts, validA, out);

  // conv1: implicit-im2col GEMM, M=65536 N=256 K=576, relu -> bf16
  gemm_wmma_kernel<1><<<dim3(1024, 4), 128, 0, stream>>>(
      cf, nullptr, bt1, c1b, fbf, nullptr, nullptr, nullptr, nullptr,
      65536, 256, 576);
  // conv2: GEMM, M=65536 N=64 K=256 -> f32 feature map [pixel][64]
  gemm_wmma_kernel<2><<<dim3(1024, 1), 128, 0, stream>>>(
      nullptr, fbf, bt2, c2b, nullptr, featmap, nullptr, nullptr, nullptr,
      65536, 64, 256);
  // grid sample 100*129 points x 64 channels -> bf16 A [100][8256]
  sample_kernel<<<12900, 64, 0, stream>>>(featmap, pts, featA);
  // feat @ w_poly^T : M=100(pad128) N=512 K=8256 -> bf16
  gemm_wmma_kernel<0><<<dim3(2, 8), 128, 0, stream>>>(
      nullptr, featA, wpolyB, nullptr, t1, nullptr, nullptr, nullptr, nullptr,
      100, 512, 8256);
  // @ w_fuse^T + b_fuse, fused coarse-poly epilogue -> d_out
  gemm_wmma_kernel<3><<<dim3(2, 4), 128, 0, stream>>>(
      nullptr, t1, wfuseB, bfuse, nullptr, nullptr, ipoly, validA, out,
      100, 256, 512);
}
